// GNNRecommender_6760278524235
// MI455X (gfx1250) — matrix operand order repaired and checked by reading, not compiled, not run on hardware
//
#include <hip/hip_runtime.h>
#include <hip/hip_bf16.h>

// ---------------- problem constants (from reference) ----------------
#define N_USER_C 100000
#define N_ITEM_C 200000
#define NEDGE_C  1600000
#define K_ITER   10
#define BATCH_C  16384
#define ALPHA_C  0.15f
#define EPS_C    1e-5f

#define ROW_LD   129   // stride == 1 mod 64 banks -> conflict-free columnar access

typedef __attribute__((ext_vector_type(16))) _Float16 v16h;
typedef __attribute__((ext_vector_type(8)))  float    v8f;

// =====================================================================
// WMMA fragment layouts (CDNA5 ISA):
//  A (16x32 f16):  lane<16: halves0-15 = K0..15 (row = lane&15)
//                  lane>=16: halves0-15 = K16..31
//  B (32x16 f16):  lane n (0-15) col n holds K0..15; lane n+16 col n holds K16..31 (order used here, as A)
//  C (16x16 f32):  VGPR v -> row v (lanes 0-15) / row v+8 (lanes 16-31), col = lane&15
// Both A and B are stored in LDS pre-swizzled so a fragment is ONE contiguous
// 32-byte (v16h) load per lane.
// =====================================================================

// A-buffer slot (in halves) for value (wave, row-in-wave m, feature c), KT fixed at 4
__device__ __forceinline__ int a_slot(int wave, int m, int c) {
  const int kt   = c >> 5, kk = c & 31;
  const int lane = m + ((kk & 16) ? 16 : 0);
  const int j    = kk & 15;
  return ((wave * 4 + kt) * 32 + lane) * 16 + j;
}

template<int KDIM, int NDIM>
__device__ __forceinline__ void wave_gemm(const v16h* __restrict__ sA,
                                          const v16h* __restrict__ sW,
                                          v8f* acc, int wave, int lane) {
  constexpr int KT = KDIM / 32;
  v16h areg[KT];
#pragma unroll
  for (int kt = 0; kt < KT; ++kt)
    areg[kt] = sA[(wave * 4 + kt) * 32 + lane];          // 32B contiguous per lane
#pragma unroll
  for (int nt = 0; nt < NDIM / 16; ++nt) {
    v8f c = {};
#pragma unroll
    for (int kt = 0; kt < KT; ++kt) {
      const v16h b = sW[(nt * KT + kt) * 32 + lane];     // 32B contiguous per lane
      c = __builtin_amdgcn_wmma_f32_16x16x32_f16(
              false, areg[kt], false, b, (short)0, c, false, false);
    }
    acc[nt] = c;
  }
}

// C-layout -> row-major f32 LDS (LN input)
template<int NDIM>
__device__ __forceinline__ void store_rows_f32(float (*sRow)[ROW_LD], const v8f* acc,
                                               const float* __restrict__ sB,
                                               int wave, int lane) {
  const int ncol  = lane & 15;
  const int mbase = wave * 16 + ((lane & 16) ? 8 : 0);
#pragma unroll
  for (int nt = 0; nt < NDIM / 16; ++nt) {
    const int col  = nt * 16 + ncol;
    const float bb = sB[col];
#pragma unroll
    for (int v = 0; v < 8; ++v)
      sRow[mbase + v][col] = acc[nt][v] + bb;
  }
}

// C-layout -> relu -> swizzled f16 A buffer (pred layer1)
__device__ __forceinline__ void store_rows_relu_swz(_Float16* __restrict__ sAh, const v8f* acc,
                                                    const float* __restrict__ sB,
                                                    int wave, int lane) {
  const int ncol = lane & 15;
  const int mb   = (lane & 16) ? 8 : 0;
#pragma unroll
  for (int nt = 0; nt < 8; ++nt) {
    const int col  = nt * 16 + ncol;
    const float bb = sB[col];
#pragma unroll
    for (int v = 0; v < 8; ++v) {
      const float t = fmaxf(acc[nt][v] + bb, 0.0f);
      sAh[a_slot(wave, mb + v, col)] = (_Float16)t;
    }
  }
}

// row-major f32 -> LayerNorm -> ELU -> swizzled f16 A buffer (one thread per row)
__device__ __forceinline__ void layernorm_elu_swz(const float (*sRow)[ROW_LD],
                                                  _Float16* __restrict__ sAh,
                                                  const float* __restrict__ g,
                                                  const float* __restrict__ be, int r) {
  float mu = 0.0f;
#pragma unroll 8
  for (int c = 0; c < 128; ++c) mu += sRow[r][c];
  mu *= (1.0f / 128.0f);
  float var = 0.0f;
#pragma unroll 8
  for (int c = 0; c < 128; ++c) { const float d = sRow[r][c] - mu; var += d * d; }
  var *= (1.0f / 128.0f);
  const float rs = __frsqrt_rn(var + EPS_C);
  const int wave = r >> 4, m = r & 15;
#pragma unroll 8
  for (int c = 0; c < 128; ++c) {
    float t = (sRow[r][c] - mu) * rs * g[c] + be[c];
    t = (t > 0.0f) ? t : (__expf(t) - 1.0f);
    sAh[a_slot(wave, m, c)] = (_Float16)t;
  }
}

// vectorized LDS weight staging (uint4 = 16B)
__device__ __forceinline__ void copy_w16(v16h* __restrict__ dst, const _Float16* __restrict__ src,
                                         int n_v16, int t, int nthr) {
  const uint4* s = (const uint4*)src;
  uint4* d = (uint4*)dst;
  for (int i = t; i < n_v16 * 2; i += nthr) d[i] = s[i];
}

// =====================================================================
// One-shot weight swizzle: f32 row-major [K,N] -> f16 WMMA-B layout
// out[((nt*KT + kt)*32 + lane)*16 + h]
// =====================================================================
__global__ void swizzle_weight_kernel(const float* __restrict__ W, _Float16* __restrict__ out,
                                      int K, int N) {
  const int i = blockIdx.x * blockDim.x + threadIdx.x;
  if (i >= K * N) return;
  const int k = i / N, n = i - k * N;
  const int kt = k >> 5, kk = k & 31;
  const int KT = K >> 5;
  const int lane = (n & 15) + ((kk & 16) ? 16 : 0);
  const int h = kk & 15;
  out[(((n >> 4) * KT + kt) * 32 + lane) * 16 + h] = (_Float16)W[i];
}

// =====================================================================
// Kernel 1: node MLP (128 ->LN/ELU-> 128 ->LN/ELU-> 64). 2 waves, 32 rows/block.
// =====================================================================
__global__ __launch_bounds__(64)
void node_mlp_kernel(const float* __restrict__ x,
                     const _Float16* __restrict__ sw1, const float* __restrict__ b1,
                     const float* __restrict__ g1, const float* __restrict__ be1,
                     const _Float16* __restrict__ sw2, const float* __restrict__ b2,
                     const float* __restrict__ g2, const float* __restrict__ be2,
                     const _Float16* __restrict__ sw3, const float* __restrict__ b3,
                     float* __restrict__ out) {
  __shared__ v16h  sW[1024];            // 32 KB swizzled weights (reused per layer)
  __shared__ v16h  sA[256];             // 8 KB swizzled A fragments (2 waves x 4 kt x 32)
  __shared__ float sRow[32][ROW_LD];    // 16.1 KB row-major f32 (LN input)
  __shared__ float sB[128];
  _Float16* sAh = (_Float16*)sA;

  const int tid  = threadIdx.x;
  const int wave = tid >> 5, lane = tid & 31;
  const long row0 = (long)blockIdx.x * 32;

  // phase 0: stage inputs (f32->f16 swizzled) + layer-1 weights/bias
  for (int i = tid; i < 32 * 128; i += 64) {
    const int r = i >> 7, c = i & 127;
    sAh[a_slot(r >> 4, r & 15, c)] = (_Float16)x[row0 * 128 + i];
  }
  copy_w16(sW, sw1, 1024, tid, 64);
  for (int i = tid; i < 128; i += 64) sB[i] = b1[i];
  __syncthreads();

  v8f acc[8];
  // layer 1
  wave_gemm<128, 128>(sA, sW, acc, wave, lane);
  store_rows_f32<128>(sRow, acc, sB, wave, lane);
  __syncthreads();
  // LN+ELU (threads 0-31) overlapped with layer-2 weight staging (threads 32-63)
  if (tid < 32) {
    layernorm_elu_swz(sRow, sAh, g1, be1, tid);
  } else {
    copy_w16(sW, sw2, 1024, tid - 32, 32);
    for (int i = tid - 32; i < 128; i += 32) sB[i] = b2[i];
  }
  __syncthreads();

  // layer 2
  wave_gemm<128, 128>(sA, sW, acc, wave, lane);
  store_rows_f32<128>(sRow, acc, sB, wave, lane);
  __syncthreads();
  if (tid < 32) {
    layernorm_elu_swz(sRow, sAh, g2, be2, tid);
  } else {
    copy_w16(sW, sw3, 512, tid - 32, 32);
    for (int i = tid - 32; i < 64; i += 32) sB[i] = b3[i];
  }
  __syncthreads();

  // layer 3: 128 -> 64, write straight to global
  wave_gemm<128, 64>(sA, sW, acc, wave, lane);
  {
    const int ncol  = lane & 15;
    const int mbase = wave * 16 + ((lane & 16) ? 8 : 0);
#pragma unroll
    for (int nt = 0; nt < 4; ++nt) {
      const float bb = sB[nt * 16 + ncol];
#pragma unroll
      for (int v = 0; v < 8; ++v)
        out[(row0 + mbase + v) * 64 + nt * 16 + ncol] = acc[nt][v] + bb;
    }
  }
}

// =====================================================================
// Kernel 2: prediction MLP: concat(zu,zi)[128] ->relu-> 128 ->relu-> 64 -> dot -> 1
// =====================================================================
__global__ __launch_bounds__(64)
void pred_kernel(const float* __restrict__ hu, const float* __restrict__ hi,
                 const int* __restrict__ uidx, const int* __restrict__ iidx,
                 const _Float16* __restrict__ sw1, const float* __restrict__ b1,
                 const _Float16* __restrict__ sw2, const float* __restrict__ b2,
                 const float* __restrict__ W3, const float* __restrict__ b3,
                 float* __restrict__ score) {
  __shared__ v16h  sW[1024];            // 32 KB
  __shared__ v16h  sA[256];             // 8 KB
  __shared__ float sOut[32][65];        // 8.3 KB layer-2 output (f32)
  __shared__ float sB[128];
  __shared__ float sW3[64];
  __shared__ int   sUI[32], sII[32];
  _Float16* sAh = (_Float16*)sA;

  const int tid  = threadIdx.x;
  const int wave = tid >> 5, lane = tid & 31;
  const long row0 = (long)blockIdx.x * 32;

  // phase 0: gather concat(zu,zi) into swizzled A, stage W1/b1
  if (tid < 32) { sUI[tid] = uidx[row0 + tid]; sII[tid] = iidx[row0 + tid]; }
  __syncthreads();
  for (int i = tid; i < 32 * 128; i += 64) {
    const int r = i >> 7, c = i & 127;
    const float v = (c < 64) ? hu[(long)sUI[r] * 64 + c]
                             : hi[(long)sII[r] * 64 + (c - 64)];
    sAh[a_slot(r >> 4, r & 15, c)] = (_Float16)v;
  }
  copy_w16(sW, sw1, 1024, tid, 64);
  for (int i = tid; i < 128; i += 64) sB[i] = b1[i];
  __syncthreads();

  v8f acc[8];
  // layer 1: 128 -> 128 relu (A reads precede A writes in wave program order)
  wave_gemm<128, 128>(sA, sW, acc, wave, lane);
  store_rows_relu_swz(sAh, acc, sB, wave, lane);
  __syncthreads();

  // phase 2: stage W2/b2/W3
  copy_w16(sW, sw2, 512, tid, 64);
  for (int i = tid; i < 64; i += 64) sB[i] = b2[i];
  if (tid < 64) sW3[tid] = W3[tid];
  __syncthreads();

  // layer 2: 128 -> 64 relu
  wave_gemm<128, 64>(sA, sW, acc, wave, lane);
  {
    const int ncol  = lane & 15;
    const int mbase = wave * 16 + ((lane & 16) ? 8 : 0);
#pragma unroll
    for (int nt = 0; nt < 4; ++nt) {
      const int col  = nt * 16 + ncol;
      const float bb = sB[col];
#pragma unroll
      for (int v = 0; v < 8; ++v)
        sOut[mbase + v][col] = fmaxf(acc[nt][v] + bb, 0.0f);
    }
  }
  __syncthreads();

  // layer 3: 64 -> 1 dot product per row
  if (tid < 32) {
    float s = b3[0];
#pragma unroll 8
    for (int c = 0; c < 64; ++c) s += sOut[tid][c] * sW3[c];
    score[row0 + tid] = s;
  }
}

// =====================================================================
// Graph propagation kernels
// =====================================================================
__global__ void zero_kernel(float* __restrict__ p, int n) {
  int i = blockIdx.x * blockDim.x + threadIdx.x;
  if (i < n) p[i] = 0.0f;
}

__global__ void count_kernel(const int* __restrict__ dst, float* __restrict__ cnt, int n) {
  int i = blockIdx.x * blockDim.x + threadIdx.x;
  if (i < n) atomicAdd(&cnt[dst[i]], 1.0f);
}

__global__ void recip_kernel(float* __restrict__ cnt, int n) {
  int i = blockIdx.x * blockDim.x + threadIdx.x;
  if (i < n) cnt[i] = 1.0f / fmaxf(cnt[i], 1.0f);
}

// 16 threads/edge, each moves one float4 (64 f32/edge); scatter-add via L2 atomics
__global__ __launch_bounds__(256)
void scatter_kernel(const float* __restrict__ h, const int* __restrict__ src,
                    const int* __restrict__ dst, float* __restrict__ agg, int nEdges) {
  const int t = blockIdx.x * blockDim.x + threadIdx.x;
  const int e = t >> 4;
  const int q = t & 15;
  if (e >= nEdges) return;
  const int s = src[e];
  const int d = dst[e];
  const float4 v = ((const float4*)(h + (long)s * 64))[q];
  float* ap = agg + (long)d * 64 + q * 4;
  atomicAdd(ap + 0, v.x);
  atomicAdd(ap + 1, v.y);
  atomicAdd(ap + 2, v.z);
  atomicAdd(ap + 3, v.w);
}

// h = (1-a)*agg/cnt + a*h0 ; also zeroes agg for the next round
__global__ void combine_kernel(float* __restrict__ h, float* __restrict__ agg,
                               const float* __restrict__ h0,
                               const float* __restrict__ rcnt, int n) {
  int i = blockIdx.x * blockDim.x + threadIdx.x;
  if (i >= n) return;
  const float a = agg[i] * rcnt[i >> 6];
  h[i] = (1.0f - ALPHA_C) * a + ALPHA_C * h0[i];
  agg[i] = 0.0f;
}

// =====================================================================
// Loss: -mean(log_sigmoid(pos - neg)) = mean(softplus(neg - pos))
// =====================================================================
__global__ __launch_bounds__(256)
void loss_kernel(const float* __restrict__ pos, const float* __restrict__ neg,
                 float* __restrict__ out, int n) {
  __shared__ float red[256];
  float s = 0.0f;
  for (int i = threadIdx.x; i < n; i += 256) {
    const float t = neg[i] - pos[i];
    const float sp = (t > 0.0f) ? (t + log1pf(__expf(-t))) : log1pf(__expf(t));
    s += sp;
  }
  red[threadIdx.x] = s;
  __syncthreads();
  for (int off = 128; off > 0; off >>= 1) {
    if (threadIdx.x < off) red[threadIdx.x] += red[threadIdx.x + off];
    __syncthreads();
  }
  if (threadIdx.x == 0) out[0] = red[0] / (float)n;
}

// =====================================================================
// Host-side orchestration
// =====================================================================
extern "C" void kernel_launch(void* const* d_in, const int* in_sizes, int n_in,
                              void* d_out, int out_size, void* d_ws, size_t ws_size,
                              hipStream_t stream) {
  const float* x_user = (const float*)d_in[0];
  const float* x_item = (const float*)d_in[1];
  const float* uW1 = (const float*)d_in[2];  const float* ub1 = (const float*)d_in[3];
  const float* ug1 = (const float*)d_in[4];  const float* ube1 = (const float*)d_in[5];
  const float* uW2 = (const float*)d_in[6];  const float* ub2 = (const float*)d_in[7];
  const float* ug2 = (const float*)d_in[8];  const float* ube2 = (const float*)d_in[9];
  const float* uW3 = (const float*)d_in[10]; const float* ub3 = (const float*)d_in[11];
  const float* iW1 = (const float*)d_in[12]; const float* ib1 = (const float*)d_in[13];
  const float* ig1 = (const float*)d_in[14]; const float* ibe1 = (const float*)d_in[15];
  const float* iW2 = (const float*)d_in[16]; const float* ib2 = (const float*)d_in[17];
  const float* ig2 = (const float*)d_in[18]; const float* ibe2 = (const float*)d_in[19];
  const float* iW3 = (const float*)d_in[20]; const float* ib3 = (const float*)d_in[21];
  const float* pW1 = (const float*)d_in[22]; const float* pb1 = (const float*)d_in[23];
  const float* pW2 = (const float*)d_in[24]; const float* pb2 = (const float*)d_in[25];
  const float* pW3 = (const float*)d_in[26]; const float* pb3 = (const float*)d_in[27];
  const int* src_ui = (const int*)d_in[28];
  const int* dst_ui = (const int*)d_in[29];
  const int* src_iu = (const int*)d_in[30];
  const int* dst_iu = (const int*)d_in[31];
  const int* user_idx = (const int*)d_in[32];
  const int* pos_idx  = (const int*)d_in[33];
  const int* neg_idx  = (const int*)d_in[34];

  char* ws = (char*)d_ws;
  size_t off = 0;
  auto carve = [&](size_t bytes) -> void* {
    void* p = (void*)(ws + off);
    off += (bytes + 255) & ~(size_t)255;
    return p;
  };
  float* h0_u  = (float*)carve((size_t)N_USER_C * 64 * 4);
  float* h0_i  = (float*)carve((size_t)N_ITEM_C * 64 * 4);
  float* h_u   = (float*)carve((size_t)N_USER_C * 64 * 4);
  float* h_i   = (float*)carve((size_t)N_ITEM_C * 64 * 4);
  float* agg_u = (float*)carve((size_t)N_USER_C * 64 * 4);
  float* agg_i = (float*)carve((size_t)N_ITEM_C * 64 * 4);
  float* cnt_u = (float*)carve((size_t)N_USER_C * 4);
  float* cnt_i = (float*)carve((size_t)N_ITEM_C * 4);
  float* pos_s = (float*)carve((size_t)BATCH_C * 4);
  float* neg_s = (float*)carve((size_t)BATCH_C * 4);
  // swizzled f16 weights
  _Float16* zUW1 = (_Float16*)carve(16384 * 2);
  _Float16* zUW2 = (_Float16*)carve(16384 * 2);
  _Float16* zUW3 = (_Float16*)carve(8192 * 2);
  _Float16* zIW1 = (_Float16*)carve(16384 * 2);
  _Float16* zIW2 = (_Float16*)carve(16384 * 2);
  _Float16* zIW3 = (_Float16*)carve(8192 * 2);
  _Float16* zPW1 = (_Float16*)carve(16384 * 2);
  _Float16* zPW2 = (_Float16*)carve(8192 * 2);

  const int ZT = 256;
  // 0) swizzle all GEMM weights into f16 WMMA-B layout (one-shot, tiny)
  swizzle_weight_kernel<<<64, ZT, 0, stream>>>(uW1, zUW1, 128, 128);
  swizzle_weight_kernel<<<64, ZT, 0, stream>>>(uW2, zUW2, 128, 128);
  swizzle_weight_kernel<<<32, ZT, 0, stream>>>(uW3, zUW3, 128, 64);
  swizzle_weight_kernel<<<64, ZT, 0, stream>>>(iW1, zIW1, 128, 128);
  swizzle_weight_kernel<<<64, ZT, 0, stream>>>(iW2, zIW2, 128, 128);
  swizzle_weight_kernel<<<32, ZT, 0, stream>>>(iW3, zIW3, 128, 64);
  swizzle_weight_kernel<<<64, ZT, 0, stream>>>(pW1, zPW1, 128, 128);
  swizzle_weight_kernel<<<32, ZT, 0, stream>>>(pW2, zPW2, 128, 64);

  // 1) node MLPs -> h0
  node_mlp_kernel<<<N_USER_C / 32, 64, 0, stream>>>(
      x_user, zUW1, ub1, ug1, ube1, zUW2, ub2, ug2, ube2, zUW3, ub3, h0_u);
  node_mlp_kernel<<<N_ITEM_C / 32, 64, 0, stream>>>(
      x_item, zIW1, ib1, ig1, ibe1, zIW2, ib2, ig2, ibe2, zIW3, ib3, h0_i);

  // 2) degree counts -> reciprocal, zero accumulators
  zero_kernel<<<(N_USER_C + ZT - 1) / ZT, ZT, 0, stream>>>(cnt_u, N_USER_C);
  zero_kernel<<<(N_ITEM_C + ZT - 1) / ZT, ZT, 0, stream>>>(cnt_i, N_ITEM_C);
  zero_kernel<<<(N_USER_C * 64 + ZT - 1) / ZT, ZT, 0, stream>>>(agg_u, N_USER_C * 64);
  zero_kernel<<<(N_ITEM_C * 64 + ZT - 1) / ZT, ZT, 0, stream>>>(agg_i, N_ITEM_C * 64);
  count_kernel<<<(NEDGE_C + ZT - 1) / ZT, ZT, 0, stream>>>(dst_ui, cnt_i, NEDGE_C);
  count_kernel<<<(NEDGE_C + ZT - 1) / ZT, ZT, 0, stream>>>(dst_iu, cnt_u, NEDGE_C);
  recip_kernel<<<(N_USER_C + ZT - 1) / ZT, ZT, 0, stream>>>(cnt_u, N_USER_C);
  recip_kernel<<<(N_ITEM_C + ZT - 1) / ZT, ZT, 0, stream>>>(cnt_i, N_ITEM_C);

  // 3) K propagation rounds (L2-resident gather/scatter)
  const float* cu = h0_u;
  const float* ci = h0_i;
  const int sblk = (NEDGE_C * 16 + 255) / 256;
  for (int k = 0; k < K_ITER; ++k) {
    scatter_kernel<<<sblk, 256, 0, stream>>>(cu, src_ui, dst_ui, agg_i, NEDGE_C);
    scatter_kernel<<<sblk, 256, 0, stream>>>(ci, src_iu, dst_iu, agg_u, NEDGE_C);
    combine_kernel<<<(N_USER_C * 64) / ZT, ZT, 0, stream>>>(h_u, agg_u, h0_u, cnt_u, N_USER_C * 64);
    combine_kernel<<<(N_ITEM_C * 64) / ZT, ZT, 0, stream>>>(h_i, agg_i, h0_i, cnt_i, N_ITEM_C * 64);
    cu = h_u;
    ci = h_i;
  }

  // 4) prediction scores + BPR loss
  pred_kernel<<<BATCH_C / 32, 64, 0, stream>>>(cu, ci, user_idx, pos_idx,
                                               zPW1, pb1, zPW2, pb2, pW3, pb3, pos_s);
  pred_kernel<<<BATCH_C / 32, 64, 0, stream>>>(cu, ci, user_idx, neg_idx,
                                               zPW1, pb1, zPW2, pb2, pW3, pb3, neg_s);
  loss_kernel<<<1, 256, 0, stream>>>(pos_s, neg_s, (float*)d_out, BATCH_C);
}
